// BertOutput_39376260170161
// MI455X (gfx1250) — 8-bit matrix operand order repaired and checked by reading, compiled, not run on hardware
//
#include <hip/hip_runtime.h>


typedef int v8i __attribute__((ext_vector_type(8)));

#define HIDDEN  1024
#define INTER   4096
#define M_TOTAL 8192
#define BLOCK_M 32
#define NWAVES  16
#define NT      4
#define KSTEP   64
#define MT_OFF  4104
#define PAIR_STRIDE 8208

__global__ __launch_bounds__(256) void quant_w_kernel(
    const float* __restrict__ W, const float* __restrict__ clip,
    signed char* __restrict__ wq, float* __restrict__ scale) {
  int n = blockIdx.x;
  const float4* row = (const float4*)(W + (long)n * INTER);
  int* wq4 = (int*)(wq + (long)n * INTER);
  float asum = 0.f;
  for (int j = threadIdx.x; j < INTER / 4; j += 256) {
    float4 v = row[j];
    int b0 = (v.x > 0.f) - (v.x < 0.f);
    int b1 = (v.y > 0.f) - (v.y < 0.f);
    int b2 = (v.z > 0.f) - (v.z < 0.f);
    int b3 = (v.w > 0.f) - (v.w < 0.f);
    wq4[j] = (b0 & 0xFF) | ((b1 & 0xFF) << 8) | ((b2 & 0xFF) << 16) | ((b3 & 0xFF) << 24);
    asum += fabsf(v.x) + fabsf(v.y) + fabsf(v.z) + fabsf(v.w);
  }
  #pragma unroll
  for (int off = 16; off > 0; off >>= 1) asum += __shfl_down(asum, off, 32);
  __shared__ float red[8];
  if ((threadIdx.x & 31) == 0) red[threadIdx.x >> 5] = asum;
  __syncthreads();
  if (threadIdx.x == 0) {
    float s = 0.f;
    #pragma unroll
    for (int i = 0; i < 8; ++i) s += red[i];
    scale[n] = clip[0] * (s * (1.0f / INTER));
  }
}

__global__ __launch_bounds__(512) void gemm_ln_kernel(
    const float* __restrict__ X,
    const signed char* __restrict__ Wq,
    const float* __restrict__ scale,
    const float* __restrict__ bias,
    const float* __restrict__ resid,
    const float* __restrict__ gamma,
    const float* __restrict__ beta,
    float* __restrict__ out) {
  __shared__ signed char As[16 * PAIR_STRIDE];
  __shared__ float s_sum[BLOCK_M];
  __shared__ float s_sq[BLOCK_M];

  const int tid = threadIdx.x;
  if (tid < BLOCK_M) { s_sum[tid] = 0.f; s_sq[tid] = 0.f; }

  const int m0 = blockIdx.x * BLOCK_M;

  {
    const float4* xs = (const float4*)(X + (long)m0 * INTER);
    #pragma unroll
    for (int it = 0; it < (BLOCK_M * INTER / 4) / 512; ++it) {
      int jf = it * 512 + tid;
      float4 v = xs[jf];
      int b0 = (v.x > 0.f) - (v.x < 0.f);
      int b1 = (v.y > 0.f) - (v.y < 0.f);
      int b2 = (v.z > 0.f) - (v.z < 0.f);
      int b3 = (v.w > 0.f) - (v.w < 0.f);
      int packed = (b0 & 0xFF) | ((b1 & 0xFF) << 8) |
                   ((b2 & 0xFF) << 16) | ((b3 & 0xFF) << 24);
      int row = jf >> 10;
      int col = (jf & 1023) << 2;
      int loff = (row & 15) * PAIR_STRIDE + (row >> 4) * MT_OFF + col;
      *(int*)&As[loff] = packed;
    }
  }
  __syncthreads();

  const int lane  = tid & 31;
  const int wave  = tid >> 5;
  const int lx    = lane & 15;
  const int hi    = lane >> 4;
  const int nbase = wave * (NT * 16);

  const signed char* aBase = &As[lx * PAIR_STRIDE + hi * 16];
  const signed char* bCol[NT];
  #pragma unroll
  for (int t = 0; t < NT; ++t)
    bCol[t] = Wq + (nbase + t * 16 + lx) * INTER + hi * 16;

  v8i acc[2][NT] = {};

  for (int k0 = 0; k0 < INTER; k0 += KSTEP) {
    v8i a[2];
    #pragma unroll
    for (int mt = 0; mt < 2; ++mt) {
      const signed char* p = aBase + mt * MT_OFF + k0;
      int2 t0 = *(const int2*)(p);
      int2 t1 = *(const int2*)(p + 8);
      int2 t2 = *(const int2*)(p + 32);
      int2 t3 = *(const int2*)(p + 40);
      a[mt][0] = t0.x; a[mt][1] = t0.y;
      a[mt][2] = t1.x; a[mt][3] = t1.y;
      a[mt][4] = t2.x; a[mt][5] = t2.y;
      a[mt][6] = t3.x; a[mt][7] = t3.y;
    }
    #pragma unroll
    for (int t = 0; t < NT; ++t) {
      int4 c0 = *(const int4*)(bCol[t] + k0);
      int4 c1 = *(const int4*)(bCol[t] + k0 + 32);
      v8i b;
      b[0] = c0.x; b[1] = c0.y; b[2] = c0.z; b[3] = c0.w;
      b[4] = c1.x; b[5] = c1.y; b[6] = c1.z; b[7] = c1.w;
      #pragma unroll
      for (int mt = 0; mt < 2; ++mt)
        acc[mt][t] = __builtin_amdgcn_wmma_i32_16x16x64_iu8(
             true, a[mt],  true, b, acc[mt][t],
             false,  false);
    }
  }

  float sc[NT], bs[NT], gm[NT], bt[NT];
  #pragma unroll
  for (int t = 0; t < NT; ++t) {
    int n = nbase + t * 16 + lx;
    sc[t] = scale[n]; bs[t] = bias[n]; gm[t] = gamma[n]; bt[t] = beta[n];
  }

  float fv[2][NT][8];
  #pragma unroll
  for (int mt = 0; mt < 2; ++mt)
    #pragma unroll
    for (int t = 0; t < NT; ++t)
      #pragma unroll
      for (int v = 0; v < 8; ++v) {
        int row = m0 + mt * 16 + v + hi * 8;
        int n   = nbase + t * 16 + lx;
        fv[mt][t][v] = sc[t] * (float)acc[mt][t][v] + bs[t] + resid[row * HIDDEN + n];
      }

  #pragma unroll
  for (int mt = 0; mt < 2; ++mt)
    #pragma unroll
    for (int v = 0; v < 8; ++v) {
      float ps = 0.f, pq = 0.f;
      #pragma unroll
      for (int t = 0; t < NT; ++t) { float x = fv[mt][t][v]; ps += x; pq += x * x; }
      #pragma unroll
      for (int off = 1; off < 16; off <<= 1) {
        ps += __shfl_xor(ps, off, 32);
        pq += __shfl_xor(pq, off, 32);
      }
      if (lx == 0) {
        int rl = mt * 16 + v + hi * 8;
        atomicAdd(&s_sum[rl], ps);
        atomicAdd(&s_sq[rl], pq);
      }
    }
  __syncthreads();

  #pragma unroll
  for (int mt = 0; mt < 2; ++mt)
    #pragma unroll
    for (int v = 0; v < 8; ++v) {
      int rl = mt * 16 + v + hi * 8;
      float mu  = s_sum[rl] * (1.0f / HIDDEN);
      float var = s_sq[rl] * (1.0f / HIDDEN) - mu * mu;
      float rstd = rsqrtf(var + 1e-12f);
      #pragma unroll
      for (int t = 0; t < NT; ++t) {
        int n = nbase + t * 16 + lx;
        out[(m0 + rl) * HIDDEN + n] = (fv[mt][t][v] - mu) * rstd * gm[t] + bt[t];
      }
    }
}

extern "C" void kernel_launch(void* const* d_in, const int* in_sizes, int n_in,
                              void* d_out, int out_size, void* d_ws, size_t ws_size,
                              hipStream_t stream) {
  const float* hs    = (const float*)d_in[0];
  const float* resid = (const float*)d_in[1];
  const float* W     = (const float*)d_in[2];
  const float* b     = (const float*)d_in[3];
  const float* clip  = (const float*)d_in[4];
  const float* gamma = (const float*)d_in[5];
  const float* beta  = (const float*)d_in[6];
  float* out = (float*)d_out;

  signed char* Wq = (signed char*)d_ws;
  float* scale    = (float*)(Wq + (size_t)HIDDEN * INTER);

  quant_w_kernel<<<HIDDEN, 256, 0, stream>>>(W, clip, Wq, scale);
  gemm_ln_kernel<<<M_TOTAL / BLOCK_M, 512, 0, stream>>>(
      hs, Wq, scale, b, resid, gamma, beta, out);
}
